// Model_4166118277693
// MI455X (gfx1250) — matrix operand order repaired and checked by reading, not compiled, not run on hardware
//
#include <hip/hip_runtime.h>

// Transformer encoder forward (B=8, L=1024, D=512, H=8, NL=6, EXP=4) for gfx1250.
// All matmuls: v_wmma_f32_16x16x32_bf16, bf16 operands (pre-transposed weights,
// bf16 activations), f32 accumulate.  GEMM A-tiles staged to LDS with the
// Tensor Data Mover (tensor_load_to_lds + s_wait_tensorcnt, double-buffered).

#define B_  8
#define L_  1024
#define D_  512
#define H_  8
#define DH_ 64
#define DF_ 2048
#define NL_ 6
#define CHUNK 256                 // K-chunk staged per TDM transfer (bf16 elems)
#define NEGMASK (-1.0e9f)

typedef __attribute__((ext_vector_type(16))) __bf16 v16bf;
typedef __attribute__((ext_vector_type(8)))  __bf16 v8bf;
typedef __attribute__((ext_vector_type(8)))  float  v8f;
typedef __attribute__((ext_vector_type(4)))  unsigned int v4u;
typedef __attribute__((ext_vector_type(8)))  int v8i;
typedef __attribute__((ext_vector_type(4)))  int v4i;

static __device__ __forceinline__ __bf16 f2bf(float f) { return (__bf16)f; }

static __device__ __forceinline__ v8f vzero8() {
  v8f z = {0.f, 0.f, 0.f, 0.f, 0.f, 0.f, 0.f, 0.f};
  return z;
}

// ---- WMMA fragment loaders (CDNA5 ISA 7.12.2, wave32) ----------------------
// A 16x32 bf16: lane half 0 -> rows 0-15, K runs [0..7] and [16..23];
// lane half 1 -> K += 8.  Two contiguous 16B loads per lane.
static __device__ __forceinline__ v16bf ldA(const __bf16* __restrict__ src,
                                            int ld, int lane) {
  const int m = lane & 15;
  const int kofs = (lane & 16) ? 8 : 0;
  const __bf16* p = src + (size_t)m * ld + kofs;
  const v8bf lo = *(const v8bf*)p;
  const v8bf hi = *(const v8bf*)(p + 16);
  v16bf a;
#pragma unroll
  for (int j = 0; j < 8; ++j) { a[j] = lo[j]; a[8 + j] = hi[j]; }
  return a;
}

// B 32x16 bf16: lane = N, K runs [0..7] and [16..23] (half 0); half 1 -> K += 8.
// Element (k,n) lives at src[n*ldN + k].  Two contiguous 16B loads per lane.
static __device__ __forceinline__ v16bf ldB(const __bf16* __restrict__ src,
                                            int ldN, int lane) {
  const int n = lane & 15;
  const int kofs = (lane & 16) ? 8 : 0;
  const __bf16* p = src + (size_t)n * ldN + kofs;
  const v8bf lo = *(const v8bf*)p;
  const v8bf hi = *(const v8bf*)(p + 16);
  v16bf b;
#pragma unroll
  for (int j = 0; j < 8; ++j) { b[j] = lo[j]; b[8 + j] = hi[j]; }
  return b;
}

// 16-lane-group reductions (C/D rows live inside one half-wave)
static __device__ __forceinline__ float redmax16(float v) {
#pragma unroll
  for (int m = 8; m; m >>= 1) v = fmaxf(v, __shfl_xor(v, m, 32));
  return v;
}
static __device__ __forceinline__ float redsum16(float v) {
#pragma unroll
  for (int m = 8; m; m >>= 1) v += __shfl_xor(v, m, 32);
  return v;
}

// ---- Tensor Data Mover: stage a 16 x CHUNK bf16 tile into LDS --------------
#if __has_builtin(__builtin_amdgcn_tensor_load_to_lds)
#define USE_TDM 1
static __device__ __forceinline__ void tdm_stage(const __bf16* gsrc, int rowStride,
                                                 __bf16* lds) {
  const unsigned long long ga = (unsigned long long)(uintptr_t)gsrc;
  const unsigned int la = (unsigned int)(uintptr_t)lds;    // LDS offset (addr[31:0])
  v4u g0;
  g0[0] = 1u;                                              // count=1, user descriptor
  g0[1] = la;                                              // lds_addr
  g0[2] = (unsigned int)ga;                                // global_addr[31:0]
  g0[3] = (unsigned int)((ga >> 32) & 0x1FFFFFFull) | (2u << 30);  // [56:32] + type=2
  const unsigned long long dim0 = (unsigned long long)CHUNK;
  const unsigned long long dim1 = 16ull;
  const unsigned long long s0 = (unsigned long long)rowStride;     // dim0 stride (elems)
  v8i g1;
  g1[0] = (int)(1u << 16);                                 // data_size=1 -> 2 bytes
  g1[1] = (int)((dim0 & 0xFFFFu) << 16);                   // tensor_dim0[15:0]
  g1[2] = (int)(((dim0 >> 16) & 0xFFFFu) | ((dim1 & 0xFFFFu) << 16));
  g1[3] = (int)(((dim1 >> 16) & 0xFFFFu) | ((unsigned int)CHUNK << 16)); // tile_dim0
  g1[4] = (int)16u;                                        // tile_dim1=16, tile_dim2=0
  g1[5] = (int)(unsigned int)s0;                           // dim0_stride[31:0]
  g1[6] = (int)((unsigned int)(s0 >> 32) & 0xFFFFu);       // dim0_stride[47:32]
  g1[7] = 0;
  v4i gz4 = {0, 0, 0, 0};
  v8i gz8 = {0, 0, 0, 0, 0, 0, 0, 0};
  // 6-arg form (amdgpu-toolchain / therock-10.0 headers):
  // (g0, g1, g2, g3, g4, cpol)
  __builtin_amdgcn_tensor_load_to_lds(g0, g1, gz4, gz4, gz8, 0);
}
#endif

// ---- positional embedding: xa = x + posemb ---------------------------------
__global__ void __launch_bounds__(256) posemb_kernel(const float* __restrict__ x,
                                                     float* __restrict__ xa) {
  size_t idx = (size_t)blockIdx.x * 256 + threadIdx.x;     // over B*L*D
  int d = (int)(idx % D_);
  int l = (int)((idx / D_) % L_);
  float fi = (float)((d < D_ / 2) ? d : (d - D_ / 2));
  float inv = __expf(fi * (-2.0f / (float)D_) * logf(10000.0f));
  float ang = (float)l * inv;
  float pe = (d < D_ / 2) ? __sinf(ang) : __cosf(ang);
  xa[idx] = x[idx] + pe;
}

// ---- weight prep: f32 [K,N] -> bf16 [N,K], per layer (blockIdx.z) ----------
__global__ void __launch_bounds__(256) wprep_kernel(const float* __restrict__ src,
                                                    __bf16* __restrict__ dst,
                                                    int K, int N) {
  const size_t sz = (size_t)K * N;
  const size_t base = (size_t)blockIdx.z * sz;
  const size_t e = (size_t)blockIdx.x * 256 + threadIdx.x;
  if (e >= sz) return;
  const int k = (int)(e / N), n = (int)(e % N);
  dst[base + (size_t)n * K + k] = f2bf(src[base + e]);
}

// ---- LayerNorm over last dim (D=512), bf16 output --------------------------
__global__ void __launch_bounds__(128) ln_kernel(const float* __restrict__ x,
                                                 const float* __restrict__ g,
                                                 const float* __restrict__ b,
                                                 __bf16* __restrict__ h) {
  __shared__ float red[2][4];
  const int row = blockIdx.x;
  const int tid = threadIdx.x;
  const int lane = tid & 31, wid = tid >> 5;
  const float4 v = ((const float4*)(x + (size_t)row * D_))[tid];
  float s  = v.x + v.y + v.z + v.w;
  float s2 = v.x * v.x + v.y * v.y + v.z * v.z + v.w * v.w;
#pragma unroll
  for (int m = 16; m; m >>= 1) { s += __shfl_xor(s, m, 32); s2 += __shfl_xor(s2, m, 32); }
  if (lane == 0) { red[0][wid] = s; red[1][wid] = s2; }
  __syncthreads();
  s  = red[0][0] + red[0][1] + red[0][2] + red[0][3];
  s2 = red[1][0] + red[1][1] + red[1][2] + red[1][3];
  const float mean = s * (1.0f / D_);
  const float var  = s2 * (1.0f / D_) - mean * mean;
  const float rstd = rsqrtf(var + 1e-3f);
  const float4 gg = ((const float4*)g)[tid];
  const float4 bb = ((const float4*)b)[tid];
  __bf16* hp = h + (size_t)row * D_ + tid * 4;
  hp[0] = f2bf((v.x - mean) * rstd * gg.x + bb.x);
  hp[1] = f2bf((v.y - mean) * rstd * gg.y + bb.y);
  hp[2] = f2bf((v.z - mean) * rstd * gg.z + bb.z);
  hp[3] = f2bf((v.w - mean) * rstd * gg.w + bb.w);
}

// ---- WMMA GEMM: 16x64 output per wave, TDM-staged A, outmode epilogue ------
// A: bf16 [M,K].  WT: bf16 [N,K] (pre-transposed weights).
// outmode 0: f32 out (+optional residual).  1: bf16 out.  2: bf16 transposed
// out Cb[(b*N + n)*L + l] (used to build Vt).  relu applies after bias.
__global__ void __launch_bounds__(32) gemm16_kernel(
    const __bf16* __restrict__ A, const __bf16* __restrict__ WT,
    const float* __restrict__ bias, const float* __restrict__ residual,
    float* __restrict__ Cf, __bf16* __restrict__ Cb,
    int M, int N, int K, int relu, int outmode) {
  const int lane = threadIdx.x;
  const int nt = blockIdx.x;                 // 64-wide N tile
  const int mt = blockIdx.y;                 // 16-wide M tile
  const __bf16* Ap = A + (size_t)mt * 16 * K;
  const __bf16* Wp = WT + (size_t)nt * 64 * K;
  v8f acc[4];
#pragma unroll
  for (int t = 0; t < 4; ++t) acc[t] = vzero8();

#if defined(USE_TDM)
  __shared__ __bf16 aStage[2][16 * CHUNK];   // 2 x 8KB double buffer
  const int nchunks = K / CHUNK;
  tdm_stage(Ap, K, aStage[0]);
  for (int kc = 0; kc < nchunks; ++kc) {
    if (kc + 1 < nchunks) {
      tdm_stage(Ap + (size_t)(kc + 1) * CHUNK, K, aStage[(kc + 1) & 1]);
      __builtin_amdgcn_s_wait_tensorcnt(1);  // current chunk resident
    } else {
      __builtin_amdgcn_s_wait_tensorcnt(0);
    }
    const __bf16* As = aStage[kc & 1];
    const __bf16* Wc = Wp + (size_t)kc * CHUNK;
#pragma unroll
    for (int kk = 0; kk < CHUNK; kk += 32) {
      __builtin_prefetch(Wc + kk + 512, 0, 1);             // global_prefetch_b8
      const v16bf a = ldA(As + kk, CHUNK, lane);
#pragma unroll
      for (int t = 0; t < 4; ++t) {
        const v16bf b = ldB(Wc + (size_t)t * 16 * K + kk, K, lane);
        acc[t] = __builtin_amdgcn_wmma_f32_16x16x32_bf16(false, a, false, b,
                                                         (short)0, acc[t], false, false);
      }
    }
  }
#else
  for (int k0 = 0; k0 < K; k0 += 32) {
    const v16bf a = ldA(Ap + k0, K, lane);
#pragma unroll
    for (int t = 0; t < 4; ++t) {
      const v16bf b = ldB(Wp + (size_t)t * 16 * K + k0, K, lane);
      acc[t] = __builtin_amdgcn_wmma_f32_16x16x32_bf16(false, a, false, b,
                                                       (short)0, acc[t], false, false);
    }
  }
#endif

  const int n = lane & 15;
  const int half = (lane >> 4) & 1;
#pragma unroll
  for (int t = 0; t < 4; ++t) {
    const int gn = nt * 64 + t * 16 + n;
    const float bv = bias ? bias[gn] : 0.f;
#pragma unroll
    for (int r = 0; r < 8; ++r) {
      const int gm = mt * 16 + r + 8 * half;
      float val = acc[t][r] + bv;
      if (relu) val = fmaxf(val, 0.f);
      if (outmode == 0) {
        if (residual) val += residual[(size_t)gm * N + gn];
        Cf[(size_t)gm * N + gn] = val;
      } else if (outmode == 1) {
        Cb[(size_t)gm * N + gn] = f2bf(val);
      } else {                                  // Vt[(b*N + gn)*L + l]
        const int bb2 = gm >> 10;               // gm / L_
        const int ll = gm & (L_ - 1);
        Cb[((size_t)bb2 * N + gn) * L_ + ll] = f2bf(val);
      }
    }
  }
}

// ---- Flash attention: one wave per (b, h, 16-query tile) -------------------
// q,k: bf16 [B*L, D] ; vt: bf16 [B, D, L] ; out: bf16 [B*L, D]
__global__ void __launch_bounds__(32) attn_kernel(const __bf16* __restrict__ q,
                                                  const __bf16* __restrict__ k,
                                                  const __bf16* __restrict__ vt,
                                                  const int* __restrict__ lengths,
                                                  __bf16* __restrict__ o) {
  __shared__ __bf16 pLds[16 * 32];             // P tile staging (1 KB)
  const int lane = threadIdx.x;
  const int qt = blockIdx.x, hh = blockIdx.y, bb = blockIdx.z;
  const int n = lane & 15;
  const int half = (lane >> 4) & 1;
  const int len = lengths[bb];

  const __bf16* qbase = q + ((size_t)bb * L_ + (size_t)qt * 16) * D_ + hh * DH_;
  const v16bf qa0 = ldA(qbase, D_, lane);            // dh 0..31
  const v16bf qa1 = ldA(qbase + 32, D_, lane);       // dh 32..63

  v8f oacc[4];
#pragma unroll
  for (int t = 0; t < 4; ++t) oacc[t] = vzero8();
  float mrun[8], srun[8];
#pragma unroll
  for (int r = 0; r < 8; ++r) { mrun[r] = -1e30f; srun[r] = 0.f; }

  for (int kc = 0; kc < L_ / 32; ++kc) {
    const __bf16* kb = k + ((size_t)bb * L_ + (size_t)kc * 32) * D_ + hh * DH_;
    v8f s0 = vzero8(), s1 = vzero8();
    {
      const v16bf b0 = ldB(kb, D_, lane);                       // keys 0-15, dh 0-31
      s0 = __builtin_amdgcn_wmma_f32_16x16x32_bf16(false, qa0, false, b0,
                                                   (short)0, s0, false, false);
      const v16bf b1 = ldB(kb + 32, D_, lane);                  // dh 32-63
      s0 = __builtin_amdgcn_wmma_f32_16x16x32_bf16(false, qa1, false, b1,
                                                   (short)0, s0, false, false);
      const v16bf c0 = ldB(kb + (size_t)16 * D_, D_, lane);     // keys 16-31
      s1 = __builtin_amdgcn_wmma_f32_16x16x32_bf16(false, qa0, false, c0,
                                                   (short)0, s1, false, false);
      const v16bf c1 = ldB(kb + (size_t)16 * D_ + 32, D_, lane);
      s1 = __builtin_amdgcn_wmma_f32_16x16x32_bf16(false, qa1, false, c1,
                                                   (short)0, s1, false, false);
    }
    const float bias0 = (kc * 32 + n      < len) ? 0.f : NEGMASK;
    const float bias1 = (kc * 32 + 16 + n < len) ? 0.f : NEGMASK;
#pragma unroll
    for (int r = 0; r < 8; ++r) {
      s0[r] = s0[r] * 0.125f + bias0;
      s1[r] = s1[r] * 0.125f + bias1;
    }
    // online softmax; row (r + 8*half) lives entirely in this lane-half
#pragma unroll
    for (int r = 0; r < 8; ++r) {
      const float cmax = fmaxf(redmax16(s0[r]), redmax16(s1[r]));
      const float mnew = fmaxf(mrun[r], cmax);
      const float scl = __expf(mrun[r] - mnew);
      const float p0 = __expf(s0[r] - mnew);
      const float p1 = __expf(s1[r] - mnew);
      srun[r] = srun[r] * scl + redsum16(p0) + redsum16(p1);
      mrun[r] = mnew;
      pLds[(r + 8 * half) * 32 + n]      = f2bf(p0);
      pLds[(r + 8 * half) * 32 + 16 + n] = f2bf(p1);
#pragma unroll
      for (int t = 0; t < 4; ++t) oacc[t][r] *= scl;
    }
    __syncthreads();
    const v16bf pa = ldA(pLds, 32, lane);                       // P as A-fragment
    const __bf16* vb = vt + ((size_t)bb * D_ + hh * DH_) * L_ + (size_t)kc * 32;
#pragma unroll
    for (int t = 0; t < 4; ++t) {
      const v16bf bB = ldB(vb + (size_t)t * 16 * L_, L_, lane); // Vt rows = dh
      oacc[t] = __builtin_amdgcn_wmma_f32_16x16x32_bf16(false, pa, false, bB,
                                                        (short)0, oacc[t], false, false);
    }
    __syncthreads();
  }
  __bf16* ob = o + ((size_t)bb * L_ + (size_t)qt * 16) * D_ + hh * DH_;
#pragma unroll
  for (int t = 0; t < 4; ++t)
#pragma unroll
    for (int r = 0; r < 8; ++r)
      ob[(size_t)(r + 8 * half) * D_ + t * 16 + n] = f2bf(oacc[t][r] / srun[r]);
}

// ---- host orchestration ----------------------------------------------------
extern "C" void kernel_launch(void* const* d_in, const int* in_sizes, int n_in,
                              void* d_out, int out_size, void* d_ws, size_t ws_size,
                              hipStream_t stream) {
  (void)in_sizes; (void)n_in; (void)out_size; (void)ws_size;
  const float* x    = (const float*)d_in[0];
  const int* lens   = (const int*)d_in[1];
  const float* Wq   = (const float*)d_in[2];
  const float* Wk   = (const float*)d_in[3];
  const float* Wv   = (const float*)d_in[4];
  const float* Wo   = (const float*)d_in[5];
  const float* ln1g = (const float*)d_in[6];
  const float* ln1b = (const float*)d_in[7];
  const float* ln2g = (const float*)d_in[8];
  const float* ln2b = (const float*)d_in[9];
  const float* w1   = (const float*)d_in[10];
  const float* b1   = (const float*)d_in[11];
  const float* w2   = (const float*)d_in[12];
  const float* b2   = (const float*)d_in[13];
  float* out = (float*)d_out;

  const size_t MT = (size_t)B_ * L_;               // 8192 token rows
  char* p = (char*)d_ws;
  float* XA = (float*)p;            p += MT * D_ * 4;          // residual stream A
  float* XB = (float*)p;            p += MT * D_ * 4;          // residual stream B
  __bf16* Hb = (__bf16*)p;          p += MT * D_ * 2;          // LN output
  __bf16* Q  = (__bf16*)p;          p += MT * D_ * 2;
  __bf16* Kb = (__bf16*)p;          p += MT * D_ * 2;
  __bf16* Vt = (__bf16*)p;          p += MT * D_ * 2;          // [B, D, L]
  __bf16* AO = (__bf16*)p;          p += MT * D_ * 2;
  __bf16* F1 = Q;                   // [MT, DF] reuses Q..AO (32 MB)
  __bf16* WTq = (__bf16*)p;         p += (size_t)NL_ * D_ * D_ * 2;
  __bf16* WTk = (__bf16*)p;         p += (size_t)NL_ * D_ * D_ * 2;
  __bf16* WTv = (__bf16*)p;         p += (size_t)NL_ * D_ * D_ * 2;
  __bf16* WTo = (__bf16*)p;         p += (size_t)NL_ * D_ * D_ * 2;
  __bf16* WT1 = (__bf16*)p;         p += (size_t)NL_ * D_ * DF_ * 2;  // [DF, D]
  __bf16* WT2 = (__bf16*)p;         p += (size_t)NL_ * DF_ * D_ * 2;  // [D, DF]

  // one-time (per call) weight transpose+convert, all layers via blockIdx.z
  wprep_kernel<<<dim3((D_ * D_) / 256, 1, NL_), 256, 0, stream>>>(Wq, WTq, D_, D_);
  wprep_kernel<<<dim3((D_ * D_) / 256, 1, NL_), 256, 0, stream>>>(Wk, WTk, D_, D_);
  wprep_kernel<<<dim3((D_ * D_) / 256, 1, NL_), 256, 0, stream>>>(Wv, WTv, D_, D_);
  wprep_kernel<<<dim3((D_ * D_) / 256, 1, NL_), 256, 0, stream>>>(Wo, WTo, D_, D_);
  wprep_kernel<<<dim3((D_ * DF_) / 256, 1, NL_), 256, 0, stream>>>(w1, WT1, D_, DF_);
  wprep_kernel<<<dim3((DF_ * D_) / 256, 1, NL_), 256, 0, stream>>>(w2, WT2, DF_, D_);

  posemb_kernel<<<(int)((MT * D_) / 256), 256, 0, stream>>>(x, XA);

  const dim3 gD(D_ / 64, (unsigned)(MT / 16));     // N=512 GEMMs
  const dim3 gF(DF_ / 64, (unsigned)(MT / 16));    // N=2048 GEMM
  for (int i = 0; i < NL_; ++i) {
    ln_kernel<<<(int)MT, 128, 0, stream>>>(XA, ln1g + (size_t)i * D_, ln1b + (size_t)i * D_, Hb);
    gemm16_kernel<<<gD, 32, 0, stream>>>(Hb, WTq + (size_t)i * D_ * D_, nullptr, nullptr,
                                         nullptr, Q,  (int)MT, D_, D_, 0, 1);
    gemm16_kernel<<<gD, 32, 0, stream>>>(Hb, WTk + (size_t)i * D_ * D_, nullptr, nullptr,
                                         nullptr, Kb, (int)MT, D_, D_, 0, 1);
    gemm16_kernel<<<gD, 32, 0, stream>>>(Hb, WTv + (size_t)i * D_ * D_, nullptr, nullptr,
                                         nullptr, Vt, (int)MT, D_, D_, 0, 2);
    attn_kernel<<<dim3(L_ / 16, H_, B_), 32, 0, stream>>>(Q, Kb, Vt, lens, AO);
    gemm16_kernel<<<gD, 32, 0, stream>>>(AO, WTo + (size_t)i * D_ * D_, nullptr, XA,
                                         XB, nullptr, (int)MT, D_, D_, 0, 0);
    ln_kernel<<<(int)MT, 128, 0, stream>>>(XB, ln2g + (size_t)i * D_, ln2b + (size_t)i * D_, Hb);
    gemm16_kernel<<<gF, 32, 0, stream>>>(Hb, WT1 + (size_t)i * D_ * DF_, b1 + (size_t)i * DF_,
                                         nullptr, nullptr, F1, (int)MT, DF_, D_, 1, 1);
    float* dst = (i == NL_ - 1) ? out : XA;
    gemm16_kernel<<<gD, 32, 0, stream>>>(F1, WT2 + (size_t)i * DF_ * D_, b2 + (size_t)i * D_,
                                         XB, dst, nullptr, (int)MT, D_, DF_, 0, 0);
  }
}
